// MultiheadAttention_8383776162582
// MI455X (gfx1250) — compile-verified, matrix operand order checked by reading, not run on hardware
//
#include <hip/hip_runtime.h>
#include <math.h>

// Problem constants (match reference)
#define E_DIM   1024
#define NHEADS  16
#define HDIM    64
#define BATCH   4
#define SEQ     2048
#define MROWS   (BATCH * SEQ)    // 8192 token rows
#define NQKV    (3 * E_DIM)      // 3072 fused output features

// GEMM blocking
#define BM 128
#define BN 128
#define BK 32
#define LDSS 40                  // padded LDS row stride (bf16 elems) = 80 B

// Attention LDS strides (bank-conflict-free padding)
#define KSTRIDE 72               // 32-key K chunk rows of 64 elems -> 144 B
#define VSTRIDE 40               // 64-row V^T chunk rows of 32 elems -> 80 B

typedef __bf16 bf16;
typedef __attribute__((ext_vector_type(16))) __bf16 bf16x16;
typedef __attribute__((ext_vector_type(8)))  __bf16 bf16x8;
typedef __attribute__((ext_vector_type(4)))  __bf16 bf16x4;
typedef __attribute__((ext_vector_type(8)))  float  f32x8;
typedef __attribute__((ext_vector_type(4)))  float  f32x4;

// ---------------------------------------------------------------------------
// WMMA: CDNA5 16x16x32 bf16, fp32 accumulate.
// A-fragment (ISA 7.12.2): lane L holds row m = L&15; lanes 0-15 hold
// K = {0..7,16..23}, lanes 16-31 hold K = {8..15,24..31}.  B-fragment: lane L
// holds column n = L&15 with the same K split (contiguous slices of B^T rows).
// ---------------------------------------------------------------------------
__device__ __forceinline__ f32x8 wmma_bf16(bf16x16 a, bf16x16 b, f32x8 c) {
  return __builtin_amdgcn_wmma_f32_16x16x32_bf16(
      false, a, false, b, (short)0, c, false, false);
}

// Load one 16x32 fragment from a row-major matrix (global or LDS).
__device__ __forceinline__ bf16x16 load_frag(const bf16* p, int ld, int row,
                                             int kBase) {
  const int lane = threadIdx.x & 31;
  const bf16* q = p + (size_t)(row + (lane & 15)) * ld + kBase + ((lane >> 4) << 3);
  bf16x8 lo = *reinterpret_cast<const bf16x8*>(q);
  bf16x8 hi = *reinterpret_cast<const bf16x8*>(q + 16);
  bf16x16 f;
#pragma unroll
  for (int i = 0; i < 8; ++i) { f[i] = lo[i]; f[i + 8] = hi[i]; }
  return f;
}

// ---------------------------------------------------------------------------
// CDNA5 async global->LDS copy (ASYNCcnt-tracked, bypasses VGPRs).
// ---------------------------------------------------------------------------
__device__ __forceinline__ void async_copy_b128(void* lds_ptr, const void* gptr) {
  unsigned int       l = (unsigned int)(uintptr_t)lds_ptr;
  unsigned long long g = (unsigned long long)(uintptr_t)gptr;
  asm volatile("global_load_async_to_lds_b128 %0, %1, off"
               :: "v"(l), "v"(g) : "memory");
}
__device__ __forceinline__ void wait_async0() {
  asm volatile("s_wait_asynccnt 0x0" ::: "memory");
}

// ---------------------------------------------------------------------------
// fp32 -> bf16 conversion, 4-wide.
// ---------------------------------------------------------------------------
__global__ void cvt_kernel(const float* __restrict__ in, bf16* __restrict__ out,
                           int n4) {
  int i = blockIdx.x * blockDim.x + threadIdx.x;
  if (i >= n4) return;
  f32x4 v = reinterpret_cast<const f32x4*>(in)[i];
  bf16x4 o;
  o[0] = (bf16)v[0]; o[1] = (bf16)v[1]; o[2] = (bf16)v[2]; o[3] = (bf16)v[3];
  reinterpret_cast<bf16x4*>(out)[i] = o;
}

// ---------------------------------------------------------------------------
// Shared GEMM mainloop: C[128x128] = A[128xK] @ B^T with B [N x K] row-major.
// 8 waves 4(M) x 2(N); each wave owns 32x64 = 2x4 accumulators.  A/B 128x32
// tiles double-buffered in LDS via async copies; one barrier per K-step.
// ---------------------------------------------------------------------------
__device__ __forceinline__ void gemm_mainloop(
    const bf16* __restrict__ A, int lda, int mBase,
    const bf16* __restrict__ Bw, int ldb, int nBase, int K,
    bf16* sA, bf16* sB,            // each [2][BM*LDSS]
    f32x8 acc[2][4]) {
  const int tid   = threadIdx.x;
  const int wv    = tid >> 5;
  const int waveM = wv & 3;
  const int waveN = wv >> 2;

  auto issue = [&](int kBase, int buf) {
#pragma unroll
    for (int i = 0; i < 2; ++i) {
      const int c   = tid + i * 256;
      const int row = c >> 2, q = c & 3;
      async_copy_b128(sA + buf * (BM * LDSS) + row * LDSS + q * 8,
                      A + (size_t)(mBase + row) * lda + kBase + q * 8);
      async_copy_b128(sB + buf * (BN * LDSS) + row * LDSS + q * 8,
                      Bw + (size_t)(nBase + row) * ldb + kBase + q * 8);
    }
  };

  issue(0, 0);
  const int nk = K / BK;
#pragma unroll 2
  for (int kt = 0; kt < nk; ++kt) {
    const int buf = kt & 1;
    wait_async0();                 // own async loads for buf landed
    __syncthreads();               // everyone's landed; prev reads finished
    if (kt + 1 < nk) issue((kt + 1) * BK, buf ^ 1);  // overlap with compute

    const bf16* At = sA + buf * (BM * LDSS);
    const bf16* Bt = sB + buf * (BN * LDSS);
    bf16x16 af[2], bf_[4];
#pragma unroll
    for (int mt = 0; mt < 2; ++mt)
      af[mt] = load_frag(At, LDSS, waveM * 32 + mt * 16, 0);
#pragma unroll
    for (int nt = 0; nt < 4; ++nt)
      bf_[nt] = load_frag(Bt, LDSS, waveN * 64 + nt * 16, 0);
#pragma unroll
    for (int mt = 0; mt < 2; ++mt)
#pragma unroll
      for (int nt = 0; nt < 4; ++nt)
        acc[mt][nt] = wmma_bf16(af[mt], bf_[nt], acc[mt][nt]);
  }
}

// ---------------------------------------------------------------------------
// Fused QKV projection as one GEMM: x[8192x1024] @ Wcat^T, Wcat = [3072x1024].
// Epilogue scatters: Q,K -> [B,H,S,D] bf16; V -> transposed [B,H,D,S] bf16.
// ---------------------------------------------------------------------------
__global__ __launch_bounds__(256) void qkv_gemm(
    const bf16* __restrict__ xb, const bf16* __restrict__ wcat,
    const float* __restrict__ bq, const float* __restrict__ bk,
    const float* __restrict__ bv,
    bf16* __restrict__ qout, bf16* __restrict__ kout, bf16* __restrict__ vtout) {
  __shared__ __attribute__((aligned(16))) bf16 sA[2][BM * LDSS];
  __shared__ __attribute__((aligned(16))) bf16 sB[2][BN * LDSS];

  const int tilesN = NQKV / BN;                 // 24
  const int mBase  = (blockIdx.x / tilesN) * BM;
  const int nBase  = (blockIdx.x % tilesN) * BN;

  f32x8 acc[2][4] = {};
  gemm_mainloop(xb, E_DIM, mBase, wcat, E_DIM, nBase, E_DIM,
                &sA[0][0], &sB[0][0], acc);

  const int lane  = threadIdx.x & 31;
  const int hi    = lane >> 4;
  const int wv    = threadIdx.x >> 5;
  const int waveM = wv & 3, waveN = wv >> 2;
#pragma unroll
  for (int mt = 0; mt < 2; ++mt) {
#pragma unroll
    for (int nt = 0; nt < 4; ++nt) {
      const int nG    = nBase + waveN * 64 + nt * 16 + (lane & 15);
      const int which = nG >> 10;               // 0=Q 1=K 2=V (tile-uniform)
      const int f     = nG & (E_DIM - 1);
      const int h     = f >> 6, d = f & (HDIM - 1);
      const float bias = (which == 0 ? bq : which == 1 ? bk : bv)[f];
#pragma unroll
      for (int r = 0; r < 8; ++r) {
        const int m = mBase + waveM * 32 + mt * 16 + r + 8 * hi;
        const int b = m >> 11, s = m & (SEQ - 1);
        const float val = acc[mt][nt][r] + bias;
        if (which < 2) {
          const size_t i = ((size_t)((b * NHEADS + h) * SEQ + s)) * HDIM + d;
          (which == 0 ? qout : kout)[i] = (bf16)val;
        } else {
          vtout[((size_t)((b * NHEADS + h) * HDIM + d)) * SEQ + s] = (bf16)val;
        }
      }
    }
  }
}

// ---------------------------------------------------------------------------
// Output projection: attn[8192x1024] @ Wo^T + bo -> fp32 d_out.
// ---------------------------------------------------------------------------
__global__ __launch_bounds__(256) void out_gemm(
    const bf16* __restrict__ ab, const bf16* __restrict__ wo,
    const float* __restrict__ bo, float* __restrict__ out) {
  __shared__ __attribute__((aligned(16))) bf16 sA[2][BM * LDSS];
  __shared__ __attribute__((aligned(16))) bf16 sB[2][BN * LDSS];

  const int tilesN = E_DIM / BN;                // 8
  const int mBase  = (blockIdx.x / tilesN) * BM;
  const int nBase  = (blockIdx.x % tilesN) * BN;

  f32x8 acc[2][4] = {};
  gemm_mainloop(ab, E_DIM, mBase, wo, E_DIM, nBase, E_DIM,
                &sA[0][0], &sB[0][0], acc);

  const int lane  = threadIdx.x & 31;
  const int hi    = lane >> 4;
  const int wv    = threadIdx.x >> 5;
  const int waveM = wv & 3, waveN = wv >> 2;
#pragma unroll
  for (int mt = 0; mt < 2; ++mt) {
#pragma unroll
    for (int nt = 0; nt < 4; ++nt) {
      const int n    = nBase + waveN * 64 + nt * 16 + (lane & 15);
      const float bs = bo[n];
#pragma unroll
      for (int r = 0; r < 8; ++r) {
        const int m = mBase + waveM * 32 + mt * 16 + r + 8 * hi;
        out[(size_t)m * E_DIM + n] = acc[mt][nt][r] + bs;
      }
    }
  }
}

// ---------------------------------------------------------------------------
// Flash attention, block-cooperative: one block = (b, h, 128 queries); the 8
// waves take 8 consecutive 16-query tiles and SHARE the 32-key K chunk
// (32x64) and V^T chunk (64x32), staged in LDS by async copies and double-
// buffered (8x traffic cut vs per-wave streaming).  Per 32-key block per
// wave: 4 score WMMAs, fp32 online softmax (shfl_xor reductions in 16-lane
// halves), P reshaped C-layout -> A-layout via a private LDS slice, 4 PV
// WMMAs into the register-resident 16x64 fp32 accumulator.
// ---------------------------------------------------------------------------
__global__ __launch_bounds__(256) void attention(
    const bf16* __restrict__ qb, const bf16* __restrict__ kb,
    const bf16* __restrict__ vtb, bf16* __restrict__ aout) {
  __shared__ __attribute__((aligned(16))) bf16 sK[2][32 * KSTRIDE];  // 9.0 KB
  __shared__ __attribute__((aligned(16))) bf16 sV[2][64 * VSTRIDE];  // 10.0 KB
  __shared__ __attribute__((aligned(16))) bf16 plds[8][16 * 32];     // 8.0 KB

  const int tid  = threadIdx.x;
  const int w    = tid >> 5;
  const int lane = tid & 31;
  const int hi   = lane >> 4;

  const int bh = blockIdx.x >> 4;                 // b*H + h
  const int qt = (blockIdx.x & 15) * 8 + w;       // this wave's 16-query tile
  const bf16* Qh = qb  + (size_t)bh * SEQ * HDIM;
  const bf16* Kh = kb  + (size_t)bh * SEQ * HDIM;
  const bf16* Vt = vtb + (size_t)bh * HDIM * SEQ; // [D, S]

  // Cooperative async staging of one 32-key chunk (2 x b128 per thread).
  auto issue = [&](int key0, int buf) {
    {  // K: 32 rows x 64 elems, thread -> (row = t>>3, q = t&7)
      const int row = tid >> 3, q = tid & 7;
      async_copy_b128(&sK[buf][row * KSTRIDE + q * 8],
                      Kh + (size_t)(key0 + row) * HDIM + q * 8);
    }
    {  // V^T: 64 rows x 32 elems, thread -> (row = t>>2, q = t&3)
      const int row = tid >> 2, q = tid & 3;
      async_copy_b128(&sV[buf][row * VSTRIDE + q * 8],
                      Vt + (size_t)row * SEQ + key0 + q * 8);
    }
  };

  // Q fragments stay in registers for the whole pass.
  const bf16x16 qf0 = load_frag(Qh, HDIM, qt * 16, 0);
  const bf16x16 qf1 = load_frag(Qh, HDIM, qt * 16, 32);

  f32x8 accO[4] = {};
  float rmax[8], rsum[8];
#pragma unroll
  for (int r = 0; r < 8; ++r) { rmax[r] = -INFINITY; rsum[r] = 0.0f; }

  const float scale = 0.125f;                     // 1/sqrt(64)
  bf16* myP = &plds[w][0];

  issue(0, 0);
  for (int kb32 = 0; kb32 < SEQ; kb32 += 32) {
    const int buf = (kb32 >> 5) & 1;
    wait_async0();                 // own async loads for buf landed
    __syncthreads();               // everyone's landed; prev reads finished
    if (kb32 + 32 < SEQ) issue(kb32 + 32, buf ^ 1);

    const bf16* Kt = sK[buf];
    const bf16* Vc = sV[buf];

    // Scores: two 16x16 tiles covering keys [kb32, kb32+32)
    f32x8 s0 = {}, s1 = {};
    {
      bf16x16 k0a = load_frag(Kt, KSTRIDE, 0,  0);
      bf16x16 k0b = load_frag(Kt, KSTRIDE, 0,  32);
      bf16x16 k1a = load_frag(Kt, KSTRIDE, 16, 0);
      bf16x16 k1b = load_frag(Kt, KSTRIDE, 16, 32);
      s0 = wmma_bf16(qf0, k0a, s0);
      s0 = wmma_bf16(qf1, k0b, s0);
      s1 = wmma_bf16(qf0, k1a, s1);
      s1 = wmma_bf16(qf1, k1b, s1);
    }

    // Online softmax per row.
#pragma unroll
    for (int r = 0; r < 8; ++r) {
      float v0 = s0[r] * scale, v1 = s1[r] * scale;
      float mx = fmaxf(v0, v1);
      mx = fmaxf(mx, __shfl_xor(mx, 1));
      mx = fmaxf(mx, __shfl_xor(mx, 2));
      mx = fmaxf(mx, __shfl_xor(mx, 4));
      mx = fmaxf(mx, __shfl_xor(mx, 8));
      const float nm    = fmaxf(rmax[r], mx);
      const float alpha = __expf(rmax[r] - nm);
      rmax[r] = nm;
      const float p0 = __expf(v0 - nm);
      const float p1 = __expf(v1 - nm);
      float ps = p0 + p1;
      ps += __shfl_xor(ps, 1);
      ps += __shfl_xor(ps, 2);
      ps += __shfl_xor(ps, 4);
      ps += __shfl_xor(ps, 8);
      rsum[r] = rsum[r] * alpha + ps;
#pragma unroll
      for (int nt = 0; nt < 4; ++nt) accO[nt][r] *= alpha;
      const int row = r + 8 * hi, col = lane & 15;
      myP[row * 32 + col]      = (bf16)p0;
      myP[row * 32 + 16 + col] = (bf16)p1;
    }

    // P (A-layout) x V-chunk (B-layout from V^T rows) -> accumulate O.
    const bf16x16 pf = load_frag(myP, 32, 0, 0);  // wave-private; DScnt orders
#pragma unroll
    for (int nt = 0; nt < 4; ++nt) {
      bf16x16 vf = load_frag(Vc, VSTRIDE, nt * 16, 0);
      accO[nt] = wmma_bf16(pf, vf, accO[nt]);
    }
  }

  // Epilogue: normalize, write bf16 attention output as [B*S, E].
  const int col = lane & 15;
  const int h = bh & (NHEADS - 1);
  const int b = bh >> 4;
#pragma unroll
  for (int r = 0; r < 8; ++r) {
    const float inv = 1.0f / rsum[r];
    const int s = qt * 16 + r + 8 * hi;
    const size_t rowBase = ((size_t)(b * SEQ + s)) * E_DIM + h * HDIM;
#pragma unroll
    for (int nt = 0; nt < 4; ++nt)
      aout[rowBase + nt * 16 + col] = (bf16)(accO[nt][r] * inv);
  }
}

// ---------------------------------------------------------------------------
// Launch: convert -> fused QKV GEMM -> attention -> out projection.
// Workspace (~88 MB): xb | Wcat(q,k,v) | Wo | Q | K | V^T | attn.
// ---------------------------------------------------------------------------
extern "C" void kernel_launch(void* const* d_in, const int* in_sizes, int n_in,
                              void* d_out, int out_size, void* d_ws, size_t ws_size,
                              hipStream_t stream) {
  const float* x  = (const float*)d_in[0];
  const float* Wq = (const float*)d_in[1];
  const float* bq = (const float*)d_in[2];
  const float* Wk = (const float*)d_in[3];
  const float* bk = (const float*)d_in[4];
  const float* Wv = (const float*)d_in[5];
  const float* bv = (const float*)d_in[6];
  const float* Wo = (const float*)d_in[7];
  const float* bo = (const float*)d_in[8];
  float* out = (float*)d_out;

  char* ws = (char*)d_ws;
  const size_t xBytes = (size_t)MROWS * E_DIM * sizeof(bf16);   // 16 MB
  const size_t wBytes = (size_t)E_DIM * E_DIM * sizeof(bf16);   // 2 MB
  bf16* xb   = (bf16*)ws; ws += xBytes;
  bf16* wcat = (bf16*)ws; ws += 3 * wBytes;   // rows: Wq | Wk | Wv
  bf16* wob  = (bf16*)ws; ws += wBytes;
  bf16* qb   = (bf16*)ws; ws += xBytes;
  bf16* kbf  = (bf16*)ws; ws += xBytes;
  bf16* vtb  = (bf16*)ws; ws += xBytes;
  bf16* abf  = (bf16*)ws; ws += xBytes;

  const int threads = 256;

  // 1) fp32 -> bf16
  {
    int n4 = (MROWS * E_DIM) / 4;
    cvt_kernel<<<n4 / threads, threads, 0, stream>>>(x, xb, n4);
    int w4 = (E_DIM * E_DIM) / 4;
    cvt_kernel<<<w4 / threads, threads, 0, stream>>>(Wq, wcat, w4);
    cvt_kernel<<<w4 / threads, threads, 0, stream>>>(Wk, wcat + (size_t)E_DIM * E_DIM, w4);
    cvt_kernel<<<w4 / threads, threads, 0, stream>>>(Wv, wcat + 2 * (size_t)E_DIM * E_DIM, w4);
    cvt_kernel<<<w4 / threads, threads, 0, stream>>>(Wo, wob, w4);
  }

  // 2) fused QKV GEMM: 64 x 24 blocks
  qkv_gemm<<<(MROWS / BM) * (NQKV / BN), threads, 0, stream>>>(
      xb, wcat, bq, bk, bv, qb, kbf, vtb);

  // 3) flash attention: one block per (b, h, 128 queries) = 1024 blocks
  attention<<<BATCH * NHEADS * (SEQ / 128), threads, 0, stream>>>(
      qb, kbf, vtb, abf);

  // 4) output projection: 64 x 8 blocks
  out_gemm<<<(MROWS / BM) * (E_DIM / BN), threads, 0, stream>>>(
      abf, wob, bo, out);
}
